// spectralModel_39376260170149
// MI455X (gfx1250) — matrix operand order repaired and checked by reading, not compiled, not run on hardware
//
#include <hip/hip_runtime.h>
#include <hip/hip_bf16.h>
#include <math.h>

// ---------------------------------------------------------------------------
// Spectral transformer forward for MI455X (gfx1250, wave32, WMMA).
// - All GEMM-shaped work on v_wmma_f32_16x16x32_f16 (f32 accumulate); complex
//   dots pack [re|im] along K=32.
// - LDS tiles are staged directly in f16 *fragment order* so every WMMA
//   operand is two ds_load_b128, no per-fragment converts.
// - DFT kernels stage global->LDS with GLOBAL_LOAD_ASYNC_TO_LDS_B32 (ASYNCcnt).
// ---------------------------------------------------------------------------

typedef __attribute__((ext_vector_type(16))) _Float16 v16h;
typedef __attribute__((ext_vector_type(8)))  float    v8f;

constexpr int BS    = 2;     // batch
constexpr int CINC  = 3;     // input channels
constexpr int Hh    = 48;
constexpr int Ww    = 48;
constexpr int OC    = 256;
constexpr int HEADS = 16;
constexpr int HD    = 16;
constexpr int WFT   = Ww/2 + 1;      // 25
constexpr int HWT   = Hh * WFT;      // 1200 spectral tokens
constexpr int VOUT  = 768;
constexpr float PIF = 3.14159265358979f;

__device__ __forceinline__ v8f wmma_f32(v16h a, v16h b, v8f c) {
  return __builtin_amdgcn_wmma_f32_16x16x32_f16(
      false, a, false, b, (short)0, c, false, false);
}

// Fragment-order LDS tile: t[32 lanes][16 halves] f16; one 32B load per lane.
__device__ __forceinline__ v16h frag_ld(const _Float16* t) {
  return *(const v16h*)(t + (threadIdx.x & 31) * 16);
}

// A-layout slot for element (m, k) of a 16x32 tile (contiguous k, as staged here):
//   lane = m + 16*k[4],  slot = k & 15
// B-layout slot for element (k, n) of a 32x16 tile:
//   lane = n + 16*k[4],  slot = k & 15

// ---- async global->LDS staging (CDNA5, tracked with ASYNCcnt) -------------
__device__ __forceinline__ void async_g2l_b32(void* lds, const void* gaddr) {
  unsigned l = (unsigned)(uintptr_t)lds;          // LDS aperture: low 32 bits
  unsigned long long g = (unsigned long long)(uintptr_t)gaddr;
  asm volatile("global_load_async_to_lds_b32 %0, %1, off"
               :: "v"(l), "v"(g) : "memory");
}
__device__ __forceinline__ void wait_asynccnt0() {
  asm volatile("s_wait_asynccnt 0x0" ::: "memory");
}

// ---------------------------------------------------------------------------
// Diagonal-order permutation tables
// ---------------------------------------------------------------------------
__global__ void k_build_perm(int* G, int* INVp) {
  if (threadIdx.x == 0 && blockIdx.x == 0) {
    int idx = 0;
    for (int s = 0; s < Hh + WFT - 1; ++s)
      for (int i = 0; i < Hh && i <= s; ++i) {
        int j = s - i;
        if (j < WFT) { int g = i * WFT + j; G[idx] = g; INVp[g] = idx; ++idx; }
      }
  }
}

// ---------------------------------------------------------------------------
// rfft2 (ortho): one 48x48 real channel per block -> 48x25 complex
// ---------------------------------------------------------------------------
__global__ void k_rfft2(const float* __restrict__ in, float* __restrict__ out) {
  __shared__ float img[Hh * Ww];
  __shared__ float tr[HWT], ti[HWT];
  __shared__ float ct[Ww], st[Ww];
  int ch = blockIdx.x;
  const float* src = in + (size_t)ch * Hh * Ww;
  for (int i = threadIdx.x; i < Hh * Ww; i += blockDim.x)
    async_g2l_b32(&img[i], src + i);
  if (threadIdx.x < Ww) {
    float a = -2.f * PIF * (float)threadIdx.x / (float)Ww;
    ct[threadIdx.x] = __cosf(a); st[threadIdx.x] = __sinf(a);
  }
  wait_asynccnt0();
  __syncthreads();
  for (int idx = threadIdx.x; idx < HWT; idx += blockDim.x) {
    int h = idx / WFT, wf = idx % WFT;
    float ar = 0.f, ai = 0.f;
    for (int w = 0; w < Ww; ++w) {
      int p = (wf * w) % Ww;
      float v = img[h * Ww + w];
      ar += v * ct[p]; ai += v * st[p];
    }
    tr[idx] = ar; ti[idx] = ai;
  }
  __syncthreads();
  const float scale = 1.f / 48.f;   // ortho: 1/sqrt(H*W)
  for (int idx = threadIdx.x; idx < HWT; idx += blockDim.x) {
    int hf = idx / WFT, wf = idx % WFT;
    float ar = 0.f, ai = 0.f;
    for (int h = 0; h < Hh; ++h) {
      int p = (hf * h) % Hh;
      float c = ct[p], s = st[p];
      float xr = tr[h * WFT + wf], xi = ti[h * WFT + wf];
      ar += xr * c - xi * s; ai += xr * s + xi * c;
    }
    size_t o = ((size_t)ch * HWT + idx) * 2;
    out[o] = ar * scale; out[o + 1] = ai * scale;
  }
}

// irfft2 (ortho): 48x25 complex -> 48x48 real (hermitian doubling along W)
__global__ void k_irfft2(const float* __restrict__ in, float* __restrict__ out) {
  __shared__ float xc[HWT * 2];        // interleaved re/im (verbatim async copy)
  __shared__ float ur[HWT], ui[HWT];
  __shared__ float ct[Ww], st[Ww];
  int ch = blockIdx.x;
  const float* src = in + (size_t)ch * HWT * 2;
  for (int i = threadIdx.x; i < HWT * 2; i += blockDim.x)
    async_g2l_b32(&xc[i], src + i);
  if (threadIdx.x < Ww) {
    float a = 2.f * PIF * (float)threadIdx.x / (float)Ww;
    ct[threadIdx.x] = __cosf(a); st[threadIdx.x] = __sinf(a);
  }
  wait_asynccnt0();
  __syncthreads();
  for (int idx = threadIdx.x; idx < HWT; idx += blockDim.x) {
    int h = idx / WFT, wf = idx % WFT;
    float ar = 0.f, ai = 0.f;
    for (int hf = 0; hf < Hh; ++hf) {
      int p = (h * hf) % Hh;
      float c = ct[p], s = st[p];
      float vr = xc[(hf * WFT + wf) * 2], vi = xc[(hf * WFT + wf) * 2 + 1];
      ar += vr * c - vi * s; ai += vr * s + vi * c;
    }
    ur[idx] = ar; ui[idx] = ai;
  }
  __syncthreads();
  const float scale = 1.f / 48.f;
  for (int idx = threadIdx.x; idx < Hh * Ww; idx += blockDim.x) {
    int h = idx / Ww, w = idx % Ww;
    float acc = 0.f;
    for (int wf = 0; wf < WFT; ++wf) {
      int p = (w * wf) % Ww;
      float alpha = (wf == 0 || wf == WFT - 1) ? 1.f : 2.f;
      acc += alpha * (ur[h * WFT + wf] * ct[p] - ui[h * WFT + wf] * st[p]);
    }
    out[(size_t)ch * Hh * Ww + idx] = acc * scale;
  }
}

// ---------------------------------------------------------------------------
// l2norm over channel dim
// ---------------------------------------------------------------------------
__global__ void k_l2norm_spatial(const float* __restrict__ in, float* __restrict__ out) {
  __shared__ float red[256];
  int pix = blockIdx.x;
  int b = pix / (Hh * Ww), p = pix % (Hh * Ww);
  size_t off = (size_t)(b * OC + threadIdx.x) * Hh * Ww + p;
  float v = in[off];
  red[threadIdx.x] = v * v;
  __syncthreads();
  for (int s = 128; s > 0; s >>= 1) {
    if ((int)threadIdx.x < s) red[threadIdx.x] += red[threadIdx.x + s];
    __syncthreads();
  }
  float n = fmaxf(sqrtf(red[0]), 1e-12f);
  out[off] = v / n;
}

__global__ void k_l2norm_tok(float* __restrict__ xio) {
  __shared__ float red[256];
  int i = blockIdx.x;
  int b = i / HWT, t = i % HWT;
  size_t off = ((size_t)(b * OC + threadIdx.x) * HWT + t) * 2;
  float re = xio[off], im = xio[off + 1];
  red[threadIdx.x] = re * re + im * im;
  __syncthreads();
  for (int s = 128; s > 0; s >>= 1) {
    if ((int)threadIdx.x < s) red[threadIdx.x] += red[threadIdx.x + s];
    __syncthreads();
  }
  float n = fmaxf(sqrtf(red[0]), 1e-12f);
  xio[off] = re / n; xio[off + 1] = im / n;
}

// ---------------------------------------------------------------------------
// Positional encoding: pe[o,h,wf] = l2norm_o(grid @ pw.T + pb); X += pe (real)
// ---------------------------------------------------------------------------
__global__ void k_pos_add(float* __restrict__ X, const float* __restrict__ pw,
                          const float* __restrict__ pb) {
  __shared__ float red[256];
  int idx = blockIdx.x;
  int h = idx / WFT, wf = idx % WFT;
  float gx = ((h < Hh / 2) ? (float)h : (float)(h - Hh)) / (float)Hh;
  float gy = (float)wf / (float)Ww;
  int o = threadIdx.x;
  float pe = gx * pw[2 * o] + gy * pw[2 * o + 1] + pb[o];
  red[o] = pe * pe;
  __syncthreads();
  for (int s = 128; s > 0; s >>= 1) {
    if (o < s) red[o] += red[o + s];
    __syncthreads();
  }
  float n = fmaxf(sqrtf(red[0]), 1e-12f);
  pe /= n;
  for (int b = 0; b < BS; ++b)
    X[(((size_t)(b * OC + o)) * HWT + idx) * 2] += pe;
}

// ---------------------------------------------------------------------------
// Complex gather / DC shift / input proj / elementwise
// ---------------------------------------------------------------------------
__global__ void k_cgather(const float* __restrict__ in, float* __restrict__ out,
                          const int* __restrict__ perm) {
  int i = blockIdx.x * blockDim.x + threadIdx.x;
  if (i >= BS * OC * HWT) return;
  int x = i % HWT, r = i / HWT;
  int g = perm[x];
  size_t s = ((size_t)r * HWT + g) * 2;
  out[2 * (size_t)i] = in[s]; out[2 * (size_t)i + 1] = in[s + 1];
}

__global__ void k_shift_dc(const float* __restrict__ flat, const float* __restrict__ ff,
                           const float* __restrict__ dcw, const float* __restrict__ dcb,
                           float* __restrict__ out) {
  int i = blockIdx.x * blockDim.x + threadIdx.x;
  if (i >= BS * OC * HWT) return;
  int t = i % HWT, o = (i / HWT) % OC, b = i / (HWT * OC);
  if (t == 0) {
    float fr = ff[2 * b], fi = ff[2 * b + 1];
    float wr = dcw[2 * o], wi = dcw[2 * o + 1];
    out[2 * (size_t)i]     = fr * wr - fi * wi + dcb[2 * o];
    out[2 * (size_t)i + 1] = fr * wi + fi * wr + dcb[2 * o + 1];
  } else {
    out[2 * (size_t)i]     = flat[2 * ((size_t)i - 1)];
    out[2 * (size_t)i + 1] = flat[2 * ((size_t)i - 1) + 1];
  }
}

__global__ void k_input_proj(const float* __restrict__ xf3, const float* __restrict__ w,
                             const float* __restrict__ bias, float* __restrict__ out) {
  int i = blockIdx.x * blockDim.x + threadIdx.x;
  if (i >= BS * OC * HWT) return;
  int t = i % HWT, o = (i / HWT) % OC, b = i / (HWT * OC);
  float ar = bias[2 * o], ai = bias[2 * o + 1];
#pragma unroll
  for (int c = 0; c < CINC; ++c) {
    size_t s = ((size_t)(b * CINC + c) * HWT + t) * 2;
    float xr = xf3[s], xi = xf3[s + 1];
    float wr = w[(o * CINC + c) * 2], wi = w[(o * CINC + c) * 2 + 1];
    ar += xr * wr - xi * wi; ai += xr * wi + xi * wr;
  }
  out[2 * (size_t)i] = ar; out[2 * (size_t)i + 1] = ai;
}

__global__ void k_acc(float* __restrict__ dst, const float* __restrict__ src, int n) {
  int i = blockIdx.x * blockDim.x + threadIdx.x;
  if (i < n) dst[i] += src[i];
}
__global__ void k_silu(float* __restrict__ out, const float* __restrict__ in, int n) {
  int i = blockIdx.x * blockDim.x + threadIdx.x;
  if (i < n) { float x = in[i]; out[i] = x / (1.f + __expf(-x)); }
}
__global__ void k_silu_add(float* __restrict__ out, const float* __restrict__ in,
                           const float* __restrict__ res, int n) {
  int i = blockIdx.x * blockDim.x + threadIdx.x;
  if (i < n) { float x = in[i]; out[i] = x / (1.f + __expf(-x)) + res[i]; }
}
__global__ void k_abs_t(const float* __restrict__ in, float* __restrict__ a) {
  int i = blockIdx.x * blockDim.x + threadIdx.x;
  if (i >= BS * OC * HWT) return;
  int t = i % HWT, c = (i / HWT) % OC, b = i / (HWT * OC);
  float re = in[2 * (size_t)i], im = in[2 * (size_t)i + 1];
  a[((size_t)(b * HWT + t)) * OC + c] = sqrtf(re * re + im * im);
}

// ---------------------------------------------------------------------------
// Complex GEMM: out[b,o,t] = sum_i in[b,i,t] * W[i,o]   (IC=OC=256, T=1200)
// Block = 4 waves sharing one token tile (A); each wave owns a 16-wide N
// slice.  Grid is exact (no divergent waves -> EXEC all ones for WMMA).
// ---------------------------------------------------------------------------
__global__ void k_cgemm(const float* __restrict__ In, const float* __restrict__ Wc,
                        const float* __restrict__ bias, float* __restrict__ Out) {
  __shared__ __align__(32) _Float16 Ah[32][16];         // shared A tile
  __shared__ __align__(32) _Float16 Brh[4][32][16];     // per-wave B (Re path)
  __shared__ __align__(32) _Float16 Bih[4][32][16];     // per-wave B (Im path)
  int wave = threadIdx.x >> 5, lane = threadIdx.x & 31;
  int blk = blockIdx.x;
  int tt  = (blk % (HWT / 16)) * 16;  blk /= (HWT / 16);
  int ng  = blk % 4;                  blk /= 4;
  int b   = blk;
  int nn0 = ng * 64 + wave * 16;
  v8f accR = {}, accI = {};
  for (int c = 0; c < OC / 16; ++c) {
    __syncthreads();
    if (c + 1 < OC / 16)
      __builtin_prefetch(In + (((size_t)(b * OC + (c + 1) * 16)) * HWT + tt) * 2, 0, 1);
    // A: 16 tokens x 16 chans complex -> fragment order [re lanes 0-15 | im lanes 16-31]
    for (int q = threadIdx.x; q < 256; q += 128) {
      int m = q >> 4, j16 = q & 15;
      const float* p = In + (((size_t)(b * OC + c * 16 + j16)) * HWT + tt + m) * 2;
      int la = m;
      int js = j16;
      Ah[la][js]      = (_Float16)p[0];
      Ah[la + 16][js] = (_Float16)p[1];
    }
    // B rows: [wr ; -wi] (Re out) and [wi ; wr] (Im out), fragment order
    for (int q = lane; q < 256; q += 32) {
      int j16 = q >> 4, n = q & 15;
      const float* p = Wc + (((size_t)(c * 16 + j16)) * OC + nn0 + n) * 2;
      float wr = p[0], wi = p[1];
      Brh[wave][n][j16]      = (_Float16)wr;
      Brh[wave][n + 16][j16] = (_Float16)(-wi);
      Bih[wave][n][j16]      = (_Float16)wi;
      Bih[wave][n + 16][j16] = (_Float16)wr;
    }
    __syncthreads();
    v16h a  = frag_ld(&Ah[0][0]);
    v16h br = frag_ld(&Brh[wave][0][0]);
    v16h bi = frag_ld(&Bih[wave][0][0]);
    accR = wmma_f32(a, br, accR);
    accI = wmma_f32(a, bi, accI);
  }
  int nn = lane & 15, mb = (lane >> 4) * 8;
  float brb = 0.f, bib = 0.f;
  if (bias) { brb = bias[2 * (nn0 + nn)]; bib = bias[2 * (nn0 + nn) + 1]; }
#pragma unroll
  for (int r = 0; r < 8; ++r) {
    int m = mb + r;
    size_t o = (((size_t)(b * OC + nn0 + nn)) * HWT + tt + m) * 2;
    Out[o] = accR[r] + brb; Out[o + 1] = accI[r] + bib;
  }
}

// ---------------------------------------------------------------------------
// Real GEMM (transposed weights): Out[m,n] = sum_k A[m,k]*Wt[n,k] + bias[n]
// ---------------------------------------------------------------------------
__global__ void k_rgemm(const float* __restrict__ A, const float* __restrict__ Wt,
                        const float* __restrict__ bias, float* __restrict__ Out,
                        int Mtot, int Ntot, int Kt) {
  __shared__ __align__(32) _Float16 Ahh[32][16];
  __shared__ __align__(32) _Float16 Bhh[4][32][16];
  int wave = threadIdx.x >> 5, lane = threadIdx.x & 31;
  int blk = blockIdx.x;
  int mt = Mtot / 16;
  int mm = (blk % mt) * 16;  blk /= mt;
  int nn0 = blk * 64 + wave * 16;
  v8f acc = {};
  for (int c = 0; c < Kt / 32; ++c) {
    __syncthreads();
    for (int q = threadIdx.x; q < 512; q += 128) {
      int m = q >> 5, k = q & 31;
      int la = m + ((k >> 4) << 4);
      int js = k & 15;
      Ahh[la][js] = (_Float16)A[(size_t)(mm + m) * Kt + c * 32 + k];
    }
    for (int q = lane; q < 512; q += 32) {
      int k = q >> 4, n = q & 15;
      Bhh[wave][n + ((k >> 4) << 4)][k & 15] =
          (_Float16)Wt[(size_t)(nn0 + n) * Kt + c * 32 + k];
    }
    __syncthreads();
    acc = wmma_f32(frag_ld(&Ahh[0][0]), frag_ld(&Bhh[wave][0][0]), acc);
  }
  int n = lane & 15, mb = (lane >> 4) * 8;
  float bb = bias ? bias[nn0 + n] : 0.f;
#pragma unroll
  for (int r = 0; r < 8; ++r)
    Out[(size_t)(mm + mb + r) * Ntot + nn0 + n] = acc[r] + bb;
}

// ---------------------------------------------------------------------------
// Flash-style complex attention.  1 wave per (b, head, 16-query tile).
// scores = |q^H k| / HD, causal mask in diagonal order, online softmax,
// out = P @ V (P real, V complex).  6 WMMAs per 32-key chunk.
// ---------------------------------------------------------------------------
__global__ void k_attn(const float* __restrict__ Q, const float* __restrict__ Kc,
                       const float* __restrict__ Vc, float* __restrict__ Out) {
  __shared__ __align__(32) _Float16 Qh[32][16];          // A frag: [qr|qi]
  __shared__ __align__(32) _Float16 SBre[2][32][16];     // score B, Re, key half
  __shared__ __align__(32) _Float16 SBim[2][32][16];     // score B, Im, key half
  __shared__ __align__(32) _Float16 VBr[32][16], VBi[32][16];
  __shared__ __align__(32) _Float16 Ph[32][16];          // probs as A frag
  __shared__ float Sr[16][32];
  __shared__ float mrow[16], lrow[16], rs[16];
  int id = blockIdx.x;
  int qt = id % (HWT / 16); id /= (HWT / 16);
  int hh = id % HEADS;      id /= HEADS;
  int bb = id;
  int lane = threadIdx.x;
  int qbase = qt * 16;
  for (int q = lane; q < 256; q += 32) {       // Q tile: k=d (re), k=16+d (im)
    int m = q >> 4, d = q & 15;
    const float* p = Q + (((size_t)(bb * OC + hh * HD + d)) * HWT + qbase + m) * 2;
    int la = m;
    int js = d;
    Qh[la][js]      = (_Float16)p[0];
    Qh[la + 16][js] = (_Float16)p[1];
  }
  if (lane < 16) { mrow[lane] = -1e30f; lrow[lane] = 0.f; }
  __syncthreads();
  v16h qa = frag_ld(&Qh[0][0]);
  v8f oR = {}, oI = {};
  int mb = (lane >> 4) * 8, nn = lane & 15;
  int nch = ((qbase + 15) >> 5) + 1;
  for (int cc = 0; cc < nch; ++cc) {
    int kb0 = cc * 32;
    __syncthreads();
    for (int q = lane; q < 512; q += 32) {
      int j = q >> 4, d = q & 15;              // j = key-in-chunk, d = hd
      int kidx = kb0 + j;
      float kr = 0.f, ki = 0.f, vr = 0.f, vi = 0.f;
      if (kidx < HWT) {
        size_t o = (((size_t)(bb * OC + hh * HD + d)) * HWT + kidx) * 2;
        kr = Kc[o]; ki = Kc[o + 1]; vr = Vc[o]; vi = Vc[o + 1];
      }
      int half = j >> 4, ncol = j & 15;
      SBre[half][ncol][d]      = (_Float16)kr;    // K rows 0..15  = kr
      SBre[half][ncol + 16][d] = (_Float16)ki;    // K rows 16..31 = ki
      SBim[half][ncol][d]      = (_Float16)ki;    // [ki ; -kr]
      SBim[half][ncol + 16][d] = (_Float16)(-kr);
      int vl = d + half * 16;                     // PV B: (Kp=j, n=d)
      VBr[vl][ncol] = (_Float16)vr;
      VBi[vl][ncol] = (_Float16)vi;
    }
    __syncthreads();
    v8f z = {};
    v8f sR0 = wmma_f32(qa, frag_ld(&SBre[0][0][0]), z);
    v8f sI0 = wmma_f32(qa, frag_ld(&SBim[0][0][0]), z);
    v8f sR1 = wmma_f32(qa, frag_ld(&SBre[1][0][0]), z);
    v8f sI1 = wmma_f32(qa, frag_ld(&SBim[1][0][0]), z);
#pragma unroll
    for (int r = 0; r < 8; ++r) {
      int m = mb + r, qidx = qbase + m;
      int k0 = kb0 + nn, k1 = kb0 + 16 + nn;
      float a0 = sqrtf(sR0[r] * sR0[r] + sI0[r] * sI0[r]) * (1.f / HD);
      float a1 = sqrtf(sR1[r] * sR1[r] + sI1[r] * sI1[r]) * (1.f / HD);
      Sr[m][nn]      = (k0 <= qidx && k0 < HWT) ? a0 : -1e30f;
      Sr[m][16 + nn] = (k1 <= qidx && k1 < HWT) ? a1 : -1e30f;
    }
    __syncthreads();
    if (lane < 16) {                             // online softmax row update
      int m = lane;
      float mx = mrow[m];
      for (int j = 0; j < 32; ++j) mx = fmaxf(mx, Sr[m][j]);
      float sc = __expf(mrow[m] - mx);
      float s = 0.f;
      for (int j = 0; j < 32; ++j) {
        float pv = __expf(Sr[m][j] - mx);
        s += pv;
        int la = m + ((j >> 4) << 4);            // write P in A-fragment order
        int js = j & 15;
        Ph[la][js] = (_Float16)pv;
      }
      lrow[m] = lrow[m] * sc + s; mrow[m] = mx; rs[m] = sc;
    }
    __syncthreads();
#pragma unroll
    for (int r = 0; r < 8; ++r) { float f = rs[mb + r]; oR[r] *= f; oI[r] *= f; }
    v16h pa = frag_ld(&Ph[0][0]);
    oR = wmma_f32(pa, frag_ld(&VBr[0][0]), oR);
    oI = wmma_f32(pa, frag_ld(&VBi[0][0]), oI);
  }
  __syncthreads();
#pragma unroll
  for (int r = 0; r < 8; ++r) {
    int m = mb + r;
    float inv = 1.f / fmaxf(lrow[m], 1e-30f);
    size_t o = (((size_t)(bb * OC + hh * HD + nn)) * HWT + qbase + m) * 2;
    Out[o] = oR[r] * inv; Out[o + 1] = oI[r] * inv;
  }
}

// ---------------------------------------------------------------------------
// Host-side orchestration
// ---------------------------------------------------------------------------
extern "C" void kernel_launch(void* const* d_in, const int* in_sizes, int n_in,
                              void* d_out, int out_size, void* d_ws, size_t ws_size,
                              hipStream_t stream) {
  (void)in_sizes; (void)n_in; (void)out_size; (void)ws_size;
  const float* x_in   = (const float*)d_in[0];
  const float* firstf = (const float*)d_in[1];
  struct Blk { const float *wq, *wk, *wv, *wo, *posw, *posb, *ffw; } blk[2];
  int bi = 2;
  for (int p = 0; p < 2; ++p) {
    blk[p].wq   = (const float*)d_in[bi + 0];
    blk[p].wk   = (const float*)d_in[bi + 1];
    blk[p].wv   = (const float*)d_in[bi + 2];
    blk[p].wo   = (const float*)d_in[bi + 3];
    blk[p].posw = (const float*)d_in[bi + 4];
    blk[p].posb = (const float*)d_in[bi + 5];
    blk[p].ffw  = (const float*)d_in[bi + 6];
    bi += 7;
  }
  const float* input_w = (const float*)d_in[16];
  const float* input_b = (const float*)d_in[17];
  const float* dc_w    = (const float*)d_in[18];
  const float* dc_b    = (const float*)d_in[19];
  const float* abs_w   = (const float*)d_in[20];
  const float* abs_b   = (const float*)d_in[21];
  const float* phase_w = (const float*)d_in[22];
  const float* phase_b = (const float*)d_in[23];
  float* dout = (float*)d_out;

  float* ws = (float*)d_ws;
  int* G    = (int*)ws;
  int* INVp = G + HWT;
  float* base = ws + 2400;
  const size_t CSZ = (size_t)BS * OC * HWT * 2;
  const size_t SSZ = (size_t)BS * OC * Hh * Ww;
  float* xf3 = base;
  float* cb0 = xf3 + (size_t)BS * CINC * HWT * 2;   // xfft (with pe)
  float* cb1 = cb0 + CSZ;                           // freqs / flat
  float* cb2 = cb1 + CSZ;                           // q
  float* cb3 = cb2 + CSZ;                           // k
  float* cb4 = cb3 + CSZ;                           // v
  float* cb5 = cb4 + CSZ;                           // attn tokens / scratch
  float* cb6 = cb5 + CSZ;                           // proj tokens / scratch
  float* sp0 = cb6 + CSZ;                           // x
  float* sp1 = sp0 + SSZ;                           // normalized
  float* sp2 = sp1 + SSZ;                           // xtf
  float* sp3 = sp2 + SSZ;                           // tmp spatial
  float* aab = sp3 + SSZ;                           // |fft| [B*HWT][OC]

  const int EW = BS * OC * HWT;
  dim3 b256(256), b128(128), b32(32);

  // ---- stem -------------------------------------------------------------
  k_build_perm<<<1, 1, 0, stream>>>(G, INVp);
  k_rfft2<<<BS * CINC, b256, 0, stream>>>(x_in, xf3);
  k_input_proj<<<(EW + 255) / 256, b256, 0, stream>>>(xf3, input_w, input_b, cb0);
  k_cgather<<<(EW + 255) / 256, b256, 0, stream>>>(cb0, cb1, G);
  k_shift_dc<<<(EW + 255) / 256, b256, 0, stream>>>(cb1, firstf, dc_w, dc_b, cb5);
  k_cgather<<<(EW + 255) / 256, b256, 0, stream>>>(cb5, cb0, INVp);
  k_irfft2<<<BS * OC, b256, 0, stream>>>(cb0, sp3);
  k_silu<<<(int)((SSZ + 255) / 256), b256, 0, stream>>>(sp0, sp3, (int)SSZ);

  const int CG_BLOCKS = BS * 4 * (HWT / 16);   // 600, exact
  // ---- transformer blocks ----------------------------------------------
  for (int p = 0; p < 2; ++p) {
    k_l2norm_spatial<<<BS * Hh * Ww, b256, 0, stream>>>(sp0, sp1);
    k_rfft2<<<BS * OC, b256, 0, stream>>>(sp1, cb0);
    k_pos_add<<<HWT, b256, 0, stream>>>(cb0, blk[p].posw, blk[p].posb);
    k_cgather<<<(EW + 255) / 256, b256, 0, stream>>>(cb0, cb1, G);
    k_cgemm<<<CG_BLOCKS, b128, 0, stream>>>(cb1, blk[p].wq, nullptr, cb2);
    k_cgemm<<<CG_BLOCKS, b128, 0, stream>>>(cb1, blk[p].wk, nullptr, cb3);
    k_cgemm<<<CG_BLOCKS, b128, 0, stream>>>(cb1, blk[p].wv, nullptr, cb4);
    k_l2norm_tok<<<BS * HWT, b256, 0, stream>>>(cb2);
    k_l2norm_tok<<<BS * HWT, b256, 0, stream>>>(cb3);
    k_l2norm_tok<<<BS * HWT, b256, 0, stream>>>(cb4);
    k_attn<<<BS * HEADS * (HWT / 16), b32, 0, stream>>>(cb2, cb3, cb4, cb5);
    k_cgemm<<<CG_BLOCKS, b128, 0, stream>>>(cb5, blk[p].wo, nullptr, cb6);
    k_cgather<<<(EW + 255) / 256, b256, 0, stream>>>(cb6, cb5, INVp);
    k_acc<<<(int)((CSZ + 255) / 256), b256, 0, stream>>>(cb5, cb0, (int)CSZ);
    k_irfft2<<<BS * OC, b256, 0, stream>>>(cb5, sp2);
    k_l2norm_spatial<<<BS * Hh * Ww, b256, 0, stream>>>(sp2, sp1);
    k_rfft2<<<BS * OC, b256, 0, stream>>>(sp1, cb0);
    k_cgemm<<<CG_BLOCKS, b128, 0, stream>>>(cb0, blk[p].ffw, nullptr, cb6);
    k_irfft2<<<BS * OC, b256, 0, stream>>>(cb6, sp3);
    k_silu_add<<<(int)((SSZ + 255) / 256), b256, 0, stream>>>(sp0, sp3, sp2, (int)SSZ);
  }

  // ---- output heads -----------------------------------------------------
  k_rfft2<<<BS * OC, b256, 0, stream>>>(sp0, cb0);
  k_abs_t<<<(EW + 255) / 256, b256, 0, stream>>>(cb0, aab);
  const int Mtot = BS * HWT;                              // 2400
  const int RG_BLOCKS = (Mtot / 16) * (VOUT / 64);        // 150*12 = 1800, exact
  k_rgemm<<<RG_BLOCKS, b128, 0, stream>>>(aab, abs_w,   abs_b,   dout, Mtot, VOUT, OC);
  k_rgemm<<<RG_BLOCKS, b128, 0, stream>>>(aab, phase_w, phase_b,
                                          dout + (size_t)Mtot * VOUT, Mtot, VOUT, OC);
}
